// RNNModel_6983616823500
// MI455X (gfx1250) — matrix operand order repaired and checked by reading, not compiled, not run on hardware
//
#include <hip/hip_runtime.h>
#include <hip/hip_bf16.h>
#include <math.h>

typedef __attribute__((ext_vector_type(16))) __bf16 v16bf;
typedef __attribute__((ext_vector_type(8)))  __bf16 v8bf;
typedef __attribute__((ext_vector_type(8)))  float  v8f;

#define B_  512
#define T_  252
#define F_  64
#define H_  512

// ---------------------------------------------------------------------------
// One-shot pack: rewrite W_x [64,512] and W_h [512,512] (fp32, row-major)
// into bf16 WMMA B-fragment layout. Fragment (kb,nb) covers K=kb*32..+31,
// N=nb*16..+15. Per lane (32 lanes), 16 contiguous bf16:
//   lane<16 : column N = nb*16+lane,    K = kb*32 + {0..7, 16..23}
//   lane>=16: column N = nb*16+lane-16, K = kb*32 + {8..15, 24..31}
// ---------------------------------------------------------------------------
__global__ void pack_weights_kernel(const float* __restrict__ Wx,
                                    const float* __restrict__ Wh,
                                    __bf16* __restrict__ dWx,
                                    __bf16* __restrict__ dWh) {
    const int WX_FRAGS = (F_ / 32) * (H_ / 16);   // 2*32  = 64
    const int WH_FRAGS = (H_ / 32) * (H_ / 16);   // 16*32 = 512
    int t = blockIdx.x * blockDim.x + threadIdx.x;
    int total = (WX_FRAGS + WH_FRAGS) * 32;
    if (t >= total) return;
    int lane = t & 31;
    int frag = t >> 5;
    const float* src;
    __bf16* dst;
    int fi;
    if (frag < WX_FRAGS) { fi = frag;            src = Wx; dst = dWx + (size_t)fi * 32 * 16; }
    else                 { fi = frag - WX_FRAGS; src = Wh; dst = dWh + (size_t)fi * 32 * 16; }
    int kb = fi >> 5;                 // fi = kb*32 + nb
    int nb = fi & 31;
    int n     = nb * 16 + (lane & 15);
    int kbase = kb * 32 + ((lane >> 4) ? 8 : 0);
    dst += lane * 16;
    for (int j = 0; j < 16; ++j)
        dst[j] = (__bf16)src[(size_t)(kbase + j + (j >= 8 ? 8 : 0)) * H_ + n];
}

// ---------------------------------------------------------------------------
// Fused RNN: one workgroup = 16 batch rows, 8 waves, wave w owns N-tiles
// w*4 .. w*4+3.  Per timestep:
//   acc[16,512] = x_t[16,64] @ Wx + h[16,512] @ Wh          (bf16 WMMA, f32 acc)
//   h = tanh(acc + b)  -> LDS (bf16)
//   out[b,t] = tanh(h . W_out + b_out)                      (LDS reduction)
// W_x fragments (64 VGPRs/lane) are kept in registers across the whole scan.
// W_h base offset is laundered per-timestep (global_load + imm offsets, no
// LICM spill), and the W_h B-fragments are explicitly DOUBLE-BUFFERED so the
// next kb-group's 4 loads are in flight while the current 4 WMMAs execute.
// ---------------------------------------------------------------------------
__global__ __launch_bounds__(256)
void rnn_fused_kernel(const float* __restrict__ x,
                      const __bf16* __restrict__ pWx,
                      const __bf16* __restrict__ pWh,
                      const float* __restrict__ bias,
                      const float* __restrict__ Wout,
                      const float* __restrict__ bout,
                      float* __restrict__ out) {
    __shared__ __align__(16) __bf16 h_lds[16 * H_];   // 16 KB
    __shared__ float wout_lds[H_];
    __shared__ float bias_lds[H_];
    __shared__ float red[16];

    const int tid    = threadIdx.x;
    const int lane   = tid & 31;
    const int wave   = tid >> 5;
    const int laneLo = lane & 15;
    const int laneHi = lane >> 4;            // 0 or 1
    const int b0     = blockIdx.x * 16;

    for (int i = tid; i < 16 * H_; i += 256) h_lds[i] = (__bf16)0.0f;
    for (int i = tid; i < H_; i += 256) { wout_lds[i] = Wout[i]; bias_lds[i] = bias[i]; }
    __syncthreads();

    const __bf16* pWxL = pWx + (size_t)lane * 16;
    const __bf16* pWhL = pWh + (size_t)lane * 16;
    const __bf16* hA   = &h_lds[(size_t)laneLo * H_ + laneHi * 8];

    // ---- Preload all W_x B-fragments this wave needs: 8 x v16bf = 64 VGPRs,
    //      held in registers for the entire scan (reused 252x). ----
    v16bf wxf[2][4];
    #pragma unroll
    for (int kb = 0; kb < 2; ++kb)
        #pragma unroll
        for (int i = 0; i < 4; ++i)
            wxf[kb][i] = *(const v16bf*)(pWxL + (size_t)(kb * 32 + wave * 4 + i) * 512);

    const float* px = x + (size_t)(b0 + laneLo) * T_ * F_;   // this lane's batch row
    const float  bo = bout[0];

    for (int t = 0; t < T_; ++t) {
        v8f acc[4];
        #pragma unroll
        for (int i = 0; i < 4; ++i)
            #pragma unroll
            for (int r = 0; r < 8; ++r) acc[i][r] = 0.0f;

        // ---- x_t @ W_x : K = 64 -> 2 k-steps of 32, weights already in regs ----
        const float* pxt = px + (size_t)t * F_;
        #pragma unroll
        for (int kb = 0; kb < 2; ++kb) {
            // A fragment (16x32 bf16): lane row = laneLo,
            // K runs {base..base+7} and {base+16..base+23}, base = kb*32 + laneHi*8
            const float* r0 = pxt + kb * 32 + laneHi * 8;
            v16bf a;
            #pragma unroll
            for (int j = 0; j < 8; ++j) {
                a[j]     = (__bf16)r0[j];
                a[j + 8] = (__bf16)r0[j + 16];
            }
            #pragma unroll
            for (int i = 0; i < 4; ++i)
                acc[i] = __builtin_amdgcn_wmma_f32_16x16x32_bf16(
                            false, a, false, wxf[kb][i], (short)0, acc[i], false, false);
        }

        // ---- h_{t-1} @ W_h : K = 512 -> 16 k-steps of 32, fully unrolled,
        //      B fragments double-buffered (prefetch kb+1 during wmma kb). ----
        unsigned zofs = 0;
        asm volatile("" : "+v"(zofs));
        const __bf16* pw = pWhL + zofs;

        v16bf bcur[4];
        #pragma unroll
        for (int i = 0; i < 4; ++i)
            bcur[i] = *(const v16bf*)(pw + (size_t)(wave * 4 + i) * 512);

        #pragma unroll
        for (int kb = 0; kb < 16; ++kb) {
            union { v16bf v; v8bf h[2]; } a;
            const __bf16* hp = hA + kb * 32;
            a.h[0] = *(const v8bf*)(hp);        // K base..base+7
            a.h[1] = *(const v8bf*)(hp + 16);   // K base+16..base+23

            v16bf bnxt[4];
            if (kb < 15) {
                #pragma unroll
                for (int i = 0; i < 4; ++i)
                    bnxt[i] = *(const v16bf*)(pw + (size_t)((kb + 1) * 32 + wave * 4 + i) * 512);
            }
            #pragma unroll
            for (int i = 0; i < 4; ++i)
                acc[i] = __builtin_amdgcn_wmma_f32_16x16x32_bf16(
                            false, a.v, false, bcur[i], (short)0, acc[i], false, false);
            if (kb < 15) {
                #pragma unroll
                for (int i = 0; i < 4; ++i) bcur[i] = bnxt[i];
            }
        }

        __syncthreads();   // everyone done reading h_lds; acc lives in registers

        // ---- h = tanh(acc + b) -> LDS ----
        #pragma unroll
        for (int i = 0; i < 4; ++i) {
            int nbG = wave * 4 + i;
            float bb = bias_lds[nbG * 16 + laneLo];
            #pragma unroll
            for (int r = 0; r < 8; ++r) {
                float v = acc[i][r] + bb;                     // M = r + laneHi*8, N = laneLo
                h_lds[(size_t)(r + laneHi * 8) * H_ + nbG * 16 + laneLo] = (__bf16)tanhf(v);
            }
        }
        if (tid < 16) red[tid] = 0.0f;
        __syncthreads();

        // ---- output head: out[b,t] = tanh(h . W_out + b_out) ----
        {
            int row = tid >> 4;            // 0..15
            int seg = tid & 15;            // 32 elements each
            const __bf16* hr = &h_lds[(size_t)row * H_ + seg * 32];
            const float*  wr = &wout_lds[seg * 32];
            float p = 0.0f;
            #pragma unroll
            for (int i = 0; i < 32; ++i) p += (float)hr[i] * wr[i];
            atomicAdd(&red[row], p);
        }
        __syncthreads();
        if (tid < 16)
            out[(size_t)(b0 + tid) * T_ + t] = tanhf(red[tid] + bo);
        // next iteration's first __syncthreads() orders red-read vs re-zero
    }
}

extern "C" void kernel_launch(void* const* d_in, const int* in_sizes, int n_in,
                              void* d_out, int out_size, void* d_ws, size_t ws_size,
                              hipStream_t stream) {
    const float* x    = (const float*)d_in[0];   // [512,252,64]
    const float* Wx   = (const float*)d_in[1];   // [64,512]
    const float* Wh   = (const float*)d_in[2];   // [512,512]
    const float* b    = (const float*)d_in[3];   // [512]
    const float* Wout = (const float*)d_in[4];   // [512,1]
    const float* bout = (const float*)d_in[5];   // [1]
    float* out = (float*)d_out;                  // [512,252,1]

    __bf16* pWx = (__bf16*)d_ws;                      // 64 frags  * 512 bf16 = 32768
    __bf16* pWh = pWx + (size_t)64 * 32 * 16;         // 512 frags * 512 bf16

    int packTotal = (64 + 512) * 32;                  // 18432 lane-tasks
    pack_weights_kernel<<<(packTotal + 255) / 256, 256, 0, stream>>>(Wx, Wh, pWx, pWh);
    rnn_fused_kernel<<<B_ / 16, 256, 0, stream>>>(x, pWx, pWh, b, Wout, bout, out);
}
